// SelfAttention_47682726920963
// MI455X (gfx1250) — compile-verified, matrix operand order checked by reading, not run on hardware
//
#include <hip/hip_runtime.h>

// ---------------------------------------------------------------------------
// Self-attention forward for MI455X (gfx1250, wave32, WMMA + TDM).
//   B=2, N=4096, D=512, H=8, Hd=64
// Pipeline:
//   1) sa_qkv_gemm  x3 : Y = x @ W^T  (fp32 -> f16 WMMA, fp32 accum)
//                        Q,K laid out [b,h,N,64]; V transposed [b,h,64,N]
//   2) sa_flash_attn   : streaming softmax(QK^T/8)V; K/V blocks DMA'd into
//                        double-buffered LDS by the Tensor Data Mover,
//                        synchronized with TENSORcnt.
//   3) sa_out_proj     : O @ Wp^T + bp -> fp32 d_out
// ---------------------------------------------------------------------------

#define D_EMBED 512
#define N_TOK   4096
#define BATCH   2
#define N_HEADS 8
#define HEAD_D  64
#define BN      (BATCH * N_TOK)                       // 8192 tokens total
#define QK_ELEMS ((size_t)BATCH * N_HEADS * N_TOK * HEAD_D)  // 4,194,304 halves
#define KBLK    32
#define NBLK    (N_TOK / KBLK)                        // 128 key blocks

typedef _Float16 v16h __attribute__((ext_vector_type(16)));
typedef _Float16 v8h  __attribute__((ext_vector_type(8)));
typedef float    v8f  __attribute__((ext_vector_type(8)));
typedef unsigned int v4u __attribute__((ext_vector_type(4)));
typedef int      v8i  __attribute__((ext_vector_type(8)));
typedef int      v4i  __attribute__((ext_vector_type(4)));

#if __has_builtin(__builtin_amdgcn_tensor_load_to_lds)
  #define HAVE_TDM 1
  #if __clang_major__ <= 22
    #define TDM_5ARG 1          // ROCm 7.2 (clang-22): 5-arg builtin
  #endif
#endif

// Per-lane fragment load for 16-bit 16x32 A (and column-major-K B) operands.
// ISA layout (CDNA5 ISA): lanes 0-15 hold K=0..7 / 16..23 in halves [0..7]/[8..15],
// lanes 16-31 hold K=8..15 / 24..31. `row` points at this lane's row (K=0).
__device__ __forceinline__ v16h load_frag(const _Float16* row, int hi) {
  const v8h lo = *(const v8h*)(row + hi * 8);
  const v8h hh = *(const v8h*)(row + 16 + hi * 8);
  v16h r;
#pragma unroll
  for (int i = 0; i < 8; ++i) { r[i] = lo[i]; r[i + 8] = hh[i]; }
  return r;
}

__device__ __forceinline__ v8f wmma_f16(v16h a, v16h b, v8f c) {
  return __builtin_amdgcn_wmma_f32_16x16x32_f16(false, a, false, b,
                                                (short)0, c, false, false);
}

#if defined(HAVE_TDM)
// Issue one TDM descriptor: 2-D tile (tile0 x tile1 elements, 2 bytes each)
// from global (row stride = stride0 elements) into contiguous LDS at lds_off.
// D# packing per CDNA5 ISA.
__device__ __forceinline__ void tdm_load_2d(unsigned lds_off, const void* gptr,
                                            unsigned tile0, unsigned tile1,
                                            unsigned tdim0, unsigned tdim1,
                                            unsigned long long stride0) {
  const unsigned long long ga = (unsigned long long)(size_t)gptr;
  v4u g0;
  g0[0] = 1u;                                          // count=1, user mode
  g0[1] = lds_off;                                     // lds_addr [63:32]
  g0[2] = (unsigned)(ga & 0xffffffffu);                // global_addr lo
  g0[3] = (unsigned)((ga >> 32) & 0x01ffffffu)         // global_addr[56:32]
        | (2u << 30);                                  // type=2 (image)
  v8i g1;
  g1[0] = (int)(1u << 16);                             // data_size=1 (2 bytes)
  g1[1] = (int)((tdim0 & 0xffffu) << 16);              // tensor_dim0[15:0]
  g1[2] = (int)(((tdim0 >> 16) & 0xffffu) | ((tdim1 & 0xffffu) << 16));
  g1[3] = (int)(((tdim1 >> 16) & 0xffffu) | ((tile0 & 0xffffu) << 16));
  g1[4] = (int)(tile1 & 0xffffu);                      // tile_dim1, tile_dim2=0
  g1[5] = (int)(stride0 & 0xffffffffu);                // dim0 stride lo
  g1[6] = (int)((stride0 >> 32) & 0xffffu);            // dim0 stride hi
  g1[7] = 0;                                           // dim1 stride unused
  const v4i z = {0, 0, 0, 0};                          // 2-D: groups 2/3 zero
#if defined(TDM_5ARG)
  __builtin_amdgcn_tensor_load_to_lds(g0, g1, z, z, 0);
#else
  const v8i z8 = {0, 0, 0, 0, 0, 0, 0, 0};
  __builtin_amdgcn_tensor_load_to_lds(g0, g1, z, z, z8, 0);
#endif
}
#endif  // HAVE_TDM

// ---------------------------------------------------------------------------
// QKV projection GEMM: Y[8192,512] = X[8192,512] @ W^T, W stored [out,in].
// mode 0: f16 out at [b,h,tok,hd]   (Q and K)
// mode 1: f16 out at [b,h,hd,tok]   (V transposed for the PV matmul)
// ---------------------------------------------------------------------------
__global__ __launch_bounds__(256) void sa_qkv_gemm(const float* __restrict__ X,
                                                   const float* __restrict__ W,
                                                   _Float16* __restrict__ out,
                                                   int mode) {
  __shared__ __align__(16) _Float16 sX[128 * 32];
  __shared__ __align__(16) _Float16 sW[64 * 32];
  const int tid  = threadIdx.x;
  const int wave = tid >> 5, lane = tid & 31;
  const int l = lane & 15, hi = (lane >> 4) & 1;
  const int mBase = blockIdx.y * 128;
  const int nBase = blockIdx.x * 64;

  v8f acc[4] = {};

  for (int kb = 0; kb < D_EMBED; kb += 32) {
#pragma unroll
    for (int j = 0; j < 4; ++j) {
      int u = tid + j * 256;
      int row = u >> 3, c4 = (u & 7) * 4;
      const float4 f = *(const float4*)(X + (size_t)(mBase + row) * D_EMBED + kb + c4);
      sX[row * 32 + c4 + 0] = (_Float16)f.x;
      sX[row * 32 + c4 + 1] = (_Float16)f.y;
      sX[row * 32 + c4 + 2] = (_Float16)f.z;
      sX[row * 32 + c4 + 3] = (_Float16)f.w;
    }
#pragma unroll
    for (int j = 0; j < 2; ++j) {
      int u = tid + j * 256;
      int row = u >> 3, c4 = (u & 7) * 4;
      const float4 f = *(const float4*)(W + (size_t)(nBase + row) * D_EMBED + kb + c4);
      sW[row * 32 + c4 + 0] = (_Float16)f.x;
      sW[row * 32 + c4 + 1] = (_Float16)f.y;
      sW[row * 32 + c4 + 2] = (_Float16)f.z;
      sW[row * 32 + c4 + 3] = (_Float16)f.w;
    }
    __syncthreads();

    const v16h a = load_frag(&sX[(wave * 16 + l) * 32], hi);
#pragma unroll
    for (int t = 0; t < 4; ++t) {
      const v16h b = load_frag(&sW[(t * 16 + l) * 32], hi);
      acc[t] = wmma_f16(a, b, acc[t]);
    }
    __syncthreads();
  }

#pragma unroll
  for (int t = 0; t < 4; ++t) {
#pragma unroll
    for (int r = 0; r < 8; ++r) {
      int row = mBase + wave * 16 + r + hi * 8;
      int col = nBase + t * 16 + l;
      int b   = row >> 12;
      int tok = row & (N_TOK - 1);
      int h   = col >> 6;
      int hd  = col & 63;
      size_t off = (mode == 0)
        ? (((size_t)(b * N_HEADS + h) * N_TOK) + tok) * HEAD_D + hd
        : (((size_t)(b * N_HEADS + h) * HEAD_D) + hd) * N_TOK + tok;
      out[off] = (_Float16)acc[t][r];
    }
  }
}

// ---------------------------------------------------------------------------
// Flash attention: 1 WG = (b,h) x 128 query rows; 8 waves x 16 rows each.
// K/V blocks DMA'd by the TDM into double-buffered LDS (TENSORcnt pipelined).
// ---------------------------------------------------------------------------
__global__ __launch_bounds__(256) void sa_flash_attn(const _Float16* __restrict__ Q,
                                                     const _Float16* __restrict__ Kf,
                                                     const _Float16* __restrict__ Vt,
                                                     _Float16* __restrict__ O) {
  __shared__ __align__(16) _Float16 sK[2][KBLK * HEAD_D];   // 2 x 4 KB
  __shared__ __align__(16) _Float16 sV[2][HEAD_D * KBLK];   // 2 x 4 KB
  __shared__ __align__(16) _Float16 sP[8 * 16 * 32];        // 8 KB
  const int tid  = threadIdx.x;
  const int wave = tid >> 5, lane = tid & 31;
  const int l = lane & 15, hi = (lane >> 4) & 1;
  const int bh    = blockIdx.y;                  // b*8 + h
  const int qBase = blockIdx.x * 128 + wave * 16;

  const _Float16* kBase = Kf + (size_t)bh * N_TOK * HEAD_D;   // [N,64] rows
  const _Float16* vBase = Vt + (size_t)bh * HEAD_D * N_TOK;   // [64,N] rows

  // Preload this wave's 16x64 Q tile as two 16x32 A fragments.
  const _Float16* qrow = Q + ((size_t)bh * N_TOK + qBase + l) * HEAD_D;
  const v16h q0 = load_frag(qrow, hi);
  const v16h q1 = load_frag(qrow + 32, hi);

  v8f o0 = {}, o1 = {}, o2 = {}, o3 = {};
  float m[8], lsum[8];
#pragma unroll
  for (int r = 0; r < 8; ++r) { m[r] = -1e30f; lsum[r] = 0.f; }

  _Float16* sPw = sP + wave * (16 * 32);
  const float scale = 0.125f;                    // 64^-0.5

#if defined(HAVE_TDM)
  const unsigned kOff0 = (unsigned)(size_t)(void*)&sK[0][0];
  const unsigned kOff1 = (unsigned)(size_t)(void*)&sK[1][0];
  const unsigned vOff0 = (unsigned)(size_t)(void*)&sV[0][0];
  const unsigned vOff1 = (unsigned)(size_t)(void*)&sV[1][0];
  // Prologue: DMA blocks 0 and 1 (K: flat 1-D 2048-half tile; V: 64x32 2-D tile).
  if (wave == 0) {
    tdm_load_2d(kOff0, kBase, KBLK * HEAD_D, 1, KBLK * HEAD_D, 1, KBLK * HEAD_D);
    tdm_load_2d(vOff0, vBase, KBLK, HEAD_D, N_TOK, HEAD_D, N_TOK);
    tdm_load_2d(kOff1, kBase + (size_t)KBLK * HEAD_D,
                KBLK * HEAD_D, 1, KBLK * HEAD_D, 1, KBLK * HEAD_D);
    tdm_load_2d(vOff1, vBase + KBLK, KBLK, HEAD_D, N_TOK, HEAD_D, N_TOK);
  }
#endif

  for (int i = 0; i < NBLK; ++i) {
    const int buf = i & 1;
    const int kb  = i * KBLK;

#if defined(HAVE_TDM)
    // Block i's two TDM ops are the oldest outstanding; in-order completion
    // means TENSORcnt<=2 (block i+1 still in flight) proves them done.
    if (wave == 0) {
      if (i + 1 < NBLK) __builtin_amdgcn_s_wait_tensorcnt(2);
      else              __builtin_amdgcn_s_wait_tensorcnt(0);
    }
    __syncthreads();
#else
    // Fallback: cooperative staging through VGPRs.
    *(v8h*)(&sK[buf][tid * 8]) = *(const v8h*)(kBase + (size_t)kb * HEAD_D + tid * 8);
    {
      int vrow = tid >> 2, c8 = (tid & 3) * 8;
      *(v8h*)(&sV[buf][vrow * 32 + c8]) =
          *(const v8h*)(vBase + (size_t)vrow * N_TOK + kb + c8);
    }
    __syncthreads();
#endif

    // S = Q K^T : two 16-key tiles, K-dim = 64 (two k-steps of 32).
    v8f s0 = {}, s1 = {};
    {
      const _Float16* kr0 = &sK[buf][l * 64];
      const _Float16* kr1 = &sK[buf][(16 + l) * 64];
      s0 = wmma_f16(q0, load_frag(kr0, hi), s0);
      s0 = wmma_f16(q1, load_frag(kr0 + 32, hi), s0);
      s1 = wmma_f16(q0, load_frag(kr1, hi), s1);
      s1 = wmma_f16(q1, load_frag(kr1 + 32, hi), s1);
    }

    // Online softmax; each C row lives in one 16-lane half => 16-wide shuffles.
#pragma unroll
    for (int r = 0; r < 8; ++r) {
      float a0 = s0[r] * scale, a1 = s1[r] * scale;
      float tmax = fmaxf(a0, a1);
#pragma unroll
      for (int off = 1; off < 16; off <<= 1)
        tmax = fmaxf(tmax, __shfl_xor(tmax, off, 16));
      float mn    = fmaxf(m[r], tmax);
      float alpha = __expf(m[r] - mn);
      m[r] = mn;
      float p0 = __expf(a0 - mn), p1 = __expf(a1 - mn);
      float rs = p0 + p1;
#pragma unroll
      for (int off = 1; off < 16; off <<= 1)
        rs += __shfl_xor(rs, off, 16);
      lsum[r] = lsum[r] * alpha + rs;
      o0[r] *= alpha; o1[r] *= alpha; o2[r] *= alpha; o3[r] *= alpha;
      sPw[(r + hi * 8) * 32 + l]      = (_Float16)p0;   // C-layout -> LDS
      sPw[(r + hi * 8) * 32 + 16 + l] = (_Float16)p1;
    }

    // O += P V : A = P tile (16x32 keys), B = rows of V^T (K-contiguous).
    const v16h ap = load_frag(sPw + l * 32, hi);
    o0 = wmma_f16(ap, load_frag(&sV[buf][l * 32], hi), o0);
    o1 = wmma_f16(ap, load_frag(&sV[buf][(16 + l) * 32], hi), o1);
    o2 = wmma_f16(ap, load_frag(&sV[buf][(32 + l) * 32], hi), o2);
    o3 = wmma_f16(ap, load_frag(&sV[buf][(48 + l) * 32], hi), o3);

    __syncthreads();   // everyone done with buf before its DMA refill

#if defined(HAVE_TDM)
    if (wave == 0 && i + 2 < NBLK) {
      const size_t kb2 = (size_t)(i + 2) * KBLK;
      tdm_load_2d(buf ? kOff1 : kOff0, kBase + kb2 * HEAD_D,
                  KBLK * HEAD_D, 1, KBLK * HEAD_D, 1, KBLK * HEAD_D);
      tdm_load_2d(buf ? vOff1 : vOff0, vBase + kb2,
                  KBLK, HEAD_D, N_TOK, HEAD_D, N_TOK);
    }
#endif
  }

  // Normalize and write O as [b, tok, h*64+hd] (head-concat layout).
  const int b = bh >> 3, h = bh & 7;
  v8f oo[4] = {o0, o1, o2, o3};
#pragma unroll
  for (int t = 0; t < 4; ++t) {
#pragma unroll
    for (int r = 0; r < 8; ++r) {
      int tok = qBase + r + hi * 8;
      int col = h * HEAD_D + t * 16 + l;
      O[((size_t)(b * N_TOK + tok)) * D_EMBED + col] = (_Float16)(oo[t][r] / lsum[r]);
    }
  }
}

// ---------------------------------------------------------------------------
// Output projection: Y = O @ Wp^T + bp, f16 input, fp32 output.
// ---------------------------------------------------------------------------
__global__ __launch_bounds__(256) void sa_out_proj(const _Float16* __restrict__ Xh,
                                                   const float* __restrict__ W,
                                                   const float* __restrict__ bias,
                                                   float* __restrict__ Y) {
  __shared__ __align__(16) _Float16 sX[128 * 32];
  __shared__ __align__(16) _Float16 sW[64 * 32];
  const int tid  = threadIdx.x;
  const int wave = tid >> 5, lane = tid & 31;
  const int l = lane & 15, hi = (lane >> 4) & 1;
  const int mBase = blockIdx.y * 128;
  const int nBase = blockIdx.x * 64;

  v8f acc[4] = {};

  for (int kb = 0; kb < D_EMBED; kb += 32) {
#pragma unroll
    for (int j = 0; j < 2; ++j) {               // X already f16: raw 16B copies
      int u = tid + j * 256;
      int row = u >> 2, c8 = (u & 3) * 8;
      *(v8h*)(sX + row * 32 + c8) =
          *(const v8h*)(Xh + (size_t)(mBase + row) * D_EMBED + kb + c8);
    }
#pragma unroll
    for (int j = 0; j < 2; ++j) {
      int u = tid + j * 256;
      int row = u >> 3, c4 = (u & 7) * 4;
      const float4 f = *(const float4*)(W + (size_t)(nBase + row) * D_EMBED + kb + c4);
      sW[row * 32 + c4 + 0] = (_Float16)f.x;
      sW[row * 32 + c4 + 1] = (_Float16)f.y;
      sW[row * 32 + c4 + 2] = (_Float16)f.z;
      sW[row * 32 + c4 + 3] = (_Float16)f.w;
    }
    __syncthreads();

    const v16h a = load_frag(&sX[(wave * 16 + l) * 32], hi);
#pragma unroll
    for (int t = 0; t < 4; ++t) {
      const v16h b = load_frag(&sW[(t * 16 + l) * 32], hi);
      acc[t] = wmma_f16(a, b, acc[t]);
    }
    __syncthreads();
  }

#pragma unroll
  for (int t = 0; t < 4; ++t) {
#pragma unroll
    for (int r = 0; r < 8; ++r) {
      int row = mBase + wave * 16 + r + hi * 8;
      int col = nBase + t * 16 + l;
      Y[(size_t)row * D_EMBED + col] = acc[t][r] + bias[col];
    }
  }
}

// ---------------------------------------------------------------------------
extern "C" void kernel_launch(void* const* d_in, const int* in_sizes, int n_in,
                              void* d_out, int out_size, void* d_ws, size_t ws_size,
                              hipStream_t stream) {
  const float* x  = (const float*)d_in[0];
  const float* Wq = (const float*)d_in[1];
  const float* Wk = (const float*)d_in[2];
  const float* Wv = (const float*)d_in[3];
  const float* Wp = (const float*)d_in[4];
  const float* bp = (const float*)d_in[5];

  _Float16* ws = (_Float16*)d_ws;       // 4 x 8 MB = 32 MB of f16 scratch
  _Float16* Qw = ws;
  _Float16* Kw = ws + QK_ELEMS;
  _Float16* Vw = ws + QK_ELEMS * 2;
  _Float16* Ow = ws + QK_ELEMS * 3;

  dim3 gemmGrid(D_EMBED / 64, BN / 128);            // (8, 64)
  sa_qkv_gemm<<<gemmGrid, 256, 0, stream>>>(x, Wq, Qw, 0);
  sa_qkv_gemm<<<gemmGrid, 256, 0, stream>>>(x, Wk, Kw, 0);
  sa_qkv_gemm<<<gemmGrid, 256, 0, stream>>>(x, Wv, Vw, 1);
  sa_flash_attn<<<dim3(N_TOK / 128, BATCH * N_HEADS), 256, 0, stream>>>(Qw, Kw, Vw, Ow);
  sa_out_proj<<<gemmGrid, 256, 0, stream>>>(Ow, Wp, bp, (float*)d_out);
}
